// MVGFRNN_no_graph_44968307589656
// MI455X (gfx1250) — matrix operand order repaired and checked by reading, not compiled, not run on hardware
//
#include <hip/hip_runtime.h>
#include <hip/hip_bf16.h>

typedef __attribute__((ext_vector_type(16))) _Float16 v16h;
typedef __attribute__((ext_vector_type(8)))  float    v8f;

#define DEV __device__ __forceinline__

// sigmoid / tanh on the native v_exp_f32 + v_rcp_f32 path (no OCML branchy
// range-split, no IEEE div_scale/div_fixup chains).  ~1 ulp, far beyond the
// f16 data precision used throughout the WMMA pipeline.
DEV float sigf(float x) {
  return __builtin_amdgcn_rcpf(1.0f + __expf(-x));
}

DEV float tanh_fast(float x) {
  float e = __expf(-2.0f * fabsf(x));
  float t = (1.0f - e) * __builtin_amdgcn_rcpf(1.0f + e);
  return copysignf(t, x);
}

DEV v8f wmma16(v16h a, v16h b, v8f c) {
  // D = A(16x32 f16) * B(32x16 f16) + C(16x16 f32)
  return __builtin_amdgcn_wmma_f32_16x16x32_f16(false, a, false, b, (short)0, c,
                                                false, false);
}

// ---- fragment helpers (layouts per CDNA5 ISA 7.12.2) -----------------------
// A (16x32 f16): lane L holds row m=L&15; element e holds K=(e<8?e:e+8)+8*(L>>4).
// B (32x16 f16): lane L holds col n=L&15; element e holds K=(e<8?e:e+8)+8*(L>>4).
// C (16x16 f32): lane L element r holds (M=r+8*(L>>4), N=L&15).

// Gather an A fragment from an LDS activation row-buffer z[m][stride] (f16,
// K contiguous), K-chunk kc covers K in [32*kc, 32*kc+32).
DEV v16h fragA_lds(const _Float16* z, int stride, int m, int half, int kc) {
  const _Float16* zr = z + m * stride + 8 * half + 32 * kc;
  v16h a;
#pragma unroll
  for (int e = 0; e < 16; ++e) { int K = (e < 8) ? e : e + 8; a[e] = zr[K]; }
  return a;
}

// Gather an A fragment straight from a global f32 row, zero-padded past Kmax.
DEV v16h fragA_glb(const float* row, int half, int kc, int Kmax) {
  v16h a;
#pragma unroll
  for (int e = 0; e < 16; ++e) {
    int K = ((e < 8) ? e : e + 8) + 8 * half + 32 * kc;
    a[e] = (K < Kmax) ? (_Float16)row[K] : (_Float16)0.0f;
  }
  return a;
}

// B fragments are prestaged in LDS so each lane reads 16 contiguous halfs.
DEV v16h fragB_lds(const _Float16* frags, int f, int lane) {
  const _Float16* p = frags + (f * 32 + lane) * 16;
  v16h b;
#pragma unroll
  for (int e = 0; e < 16; ++e) b[e] = p[e];
  return b;
}

// Stage W (Nout x Kin, row-major, f32) [optionally concat W2 (Nout x K2) along
// K] into B-fragment order.  Frag index f = nt*n_kc + kc.
DEV void stage_frags(_Float16* dst, const float* W, const float* W2, int Kin,
                     int K2, int n_nt, int n_kc, int tid, int nthr) {
  int total = n_nt * n_kc * 512;
  for (int idx = tid; idx < total; idx += nthr) {
    int e = idx & 15, ln = (idx >> 4) & 31, f = idx >> 9;
    int kc = f % n_kc, nt = f / n_kc;
    int K = ((e < 8) ? e : e + 8) + 8 * (ln >> 4) + 32 * kc;
    int n = nt * 16 + (ln & 15);
    float v = 0.f;
    if (K < Kin) v = W[n * Kin + K];
    else if (W2 && K < Kin + K2) v = W2[n * K2 + (K - Kin)];
    dst[idx] = (_Float16)v;
  }
}

template <int NKC>
DEV v8f gemm_tile(const _Float16* frags, const v16h* A, int nt, float bias,
                  int lane) {
  v8f c;
#pragma unroll
  for (int r = 0; r < 8; ++r) c[r] = bias;  // bias folded into accumulator
#pragma unroll
  for (int kc = 0; kc < NKC; ++kc)
    c = wmma16(A[kc], fragB_lds(frags, nt * NKC + kc, lane), c);
  return c;
}

// Same but with the (loop-invariant) B fragments held in registers.
DEV v8f gemm2_reg(const v16h* Bf, const v16h* A, int nt, float bias) {
  v8f c;
#pragma unroll
  for (int r = 0; r < 8; ++r) c[r] = bias;
  c = wmma16(A[0], Bf[nt * 2 + 0], c);
  c = wmma16(A[1], Bf[nt * 2 + 1], c);
  return c;
}

// ============================================================================
// Kernel 1: LSTM, last hidden state.  One wave owns 16 rows; h/c resident
// across all T steps.  z-stage row = [x_t (D) | h (32) | 0-pad] (Kpad=64).
// ovi != nullptr -> labeled variant (D=7: [ovi, meo]); else D=6 ([meo]).
// Weight fragments live in 128 VGPRs for the whole T loop.
// ============================================================================
__global__ __launch_bounds__(256) void lstm_wmma(
    const float* __restrict__ ovi, const float* __restrict__ meo,
    const float* __restrict__ Wih, const float* __restrict__ Whh,
    const float* __restrict__ bih, const float* __restrict__ bhh,
    float* __restrict__ h_out, int D, int T) {
  __shared__ _Float16 s_fragB[16 * 512];   // Wpad[64][128] as 8nt x 2kc frags
  __shared__ _Float16 s_z[8 * 16 * 64];    // per-wave activation stage
  const int tid = threadIdx.x, wave = tid >> 5, lane = tid & 31;
  const int half = lane >> 4, lm = lane & 15;
  _Float16* z = s_z + wave * (16 * 64);

  stage_frags(s_fragB, Wih, Whh, D, 32, /*n_nt=*/8, /*n_kc=*/2, tid, 256);
  for (int i = lane; i < 16 * 64; i += 32) z[i] = (_Float16)0.0f;  // h0=0, pad=0

  float biasv[8];
#pragma unroll
  for (int nt = 0; nt < 8; ++nt) biasv[nt] = bih[nt * 16 + lm] + bhh[nt * 16 + lm];

  float c_reg[16];
#pragma unroll
  for (int i = 0; i < 16; ++i) c_reg[i] = 0.f;

  const int R0 = (blockIdx.x * 8 + wave) * 16;
  __syncthreads();

  // hoist all 16 weight fragments into registers for the whole recurrence
  v16h Bf[16];
#pragma unroll
  for (int f = 0; f < 16; ++f) Bf[f] = fragB_lds(s_fragB, f, lane);

  for (int t = 0; t < T; ++t) {
    if (lane < 16) {  // stage x_t for our 16 rows
      int base = (R0 + lane) * T + t;
      _Float16* zr = z + lane * 64;
      const float* mp = meo + base * 6;
      if (ovi) {
        zr[0] = (_Float16)ovi[base];
#pragma unroll
        for (int m = 0; m < 6; ++m) zr[1 + m] = (_Float16)mp[m];
      } else {
#pragma unroll
        for (int m = 0; m < 6; ++m) zr[m] = (_Float16)mp[m];
      }
    }
    __syncthreads();  // x_t stores + previous h stores -> visible for gather

    v16h A[2];
    A[0] = fragA_lds(z, 64, lm, half, 0);
    A[1] = fragA_lds(z, 64, lm, half, 1);

#pragma unroll
    for (int ntb = 0; ntb < 2; ++ntb) {
      // PyTorch gate order i,f,g,o = gate cols [0,32),[32,64),[64,96),[96,128)
      v8f ci = gemm2_reg(Bf, A, ntb,     biasv[ntb]);
      v8f cf = gemm2_reg(Bf, A, 2 + ntb, biasv[2 + ntb]);
      v8f cg = gemm2_reg(Bf, A, 4 + ntb, biasv[4 + ntb]);
      v8f co = gemm2_reg(Bf, A, 6 + ntb, biasv[6 + ntb]);
      int j = lm + 16 * ntb;
#pragma unroll
      for (int r = 0; r < 8; ++r) {
        int m = r + 8 * half, s = ntb * 8 + r;
        float cv = sigf(cf[r]) * c_reg[s] + sigf(ci[r]) * tanh_fast(cg[r]);
        float hv = sigf(co[r]) * tanh_fast(cv);
        c_reg[s] = cv;
        z[m * 64 + D + j] = (_Float16)hv;  // recurrent h for next step's A
      }
    }
  }

  // read the final h back from our own LDS stage (same-lane, in-order)
#pragma unroll
  for (int ntb = 0; ntb < 2; ++ntb) {
    int j = lm + 16 * ntb;
#pragma unroll
    for (int r = 0; r < 8; ++r) {
      int m = r + 8 * half;
      h_out[(R0 + m) * 32 + j] = (float)z[m * 64 + D + j];
    }
  }
}

// ============================================================================
// Kernel 2: unlabeled feature branch (tiny; VALU).
// unlabel = relu([u_time | relu(feat@W_ul1^T+b)] @ W_ul2^T + b)
// ============================================================================
__global__ __launch_bounds__(64) void unlabeled_branch(
    const float* __restrict__ feat, const float* __restrict__ u_time,
    const float* __restrict__ W_ul1, const float* __restrict__ b_ul1,
    const float* __restrict__ W_ul2, const float* __restrict__ b_ul2,
    float* __restrict__ unlab) {
  __shared__ float z[64];
  int b = blockIdx.x, j = threadIdx.x;
  if (j < 32) {
    z[j] = u_time[b * 32 + j];
  } else {
    int jj = j - 32;
    const float* fr = feat + (long)b * 144;
    const float* wr = W_ul1 + jj * 144;
    float s = b_ul1[jj];
    for (int k = 0; k < 144; ++k) s += fr[k] * wr[k];
    z[32 + jj] = fmaxf(s, 0.f);
  }
  __syncthreads();
  const float* wr2 = W_ul2 + j * 64;
  float s = b_ul2[j];
#pragma unroll 8
  for (int k = 0; k < 64; ++k) s += z[k] * wr2[k];
  unlab[(long)b * 64 + j] = fmaxf(s, 0.f);
}

// ============================================================================
// Kernel 3: labeled fusion (all WMMA): l_fea -> label -> attention scalar a.
// 4 waves/block, 16 rows/wave.
// ============================================================================
__global__ __launch_bounds__(128) void labeled_fusion(
    const float* __restrict__ feat, const float* __restrict__ l_time,
    const float* __restrict__ unlab,
    const float* __restrict__ W_ll1, const float* __restrict__ b_ll1,
    const float* __restrict__ W_ll2, const float* __restrict__ b_ll2,
    const float* __restrict__ W_a1,  const float* __restrict__ b_a1,
    const float* __restrict__ W_a2,  const float* __restrict__ b_a2,
    float* __restrict__ label_out, float* __restrict__ a_out) {
  __shared__ _Float16 s_f1[10 * 512];      // W_ll1: K=145->160 (5kc) x 2nt
  __shared__ _Float16 s_f2[8 * 512];       // W_ll2: K=64 (2kc) x 4nt
  __shared__ _Float16 s_fa[4 * 512];       // W_a1 : K=128 (4kc) x 1nt
  __shared__ _Float16 s_z2[4 * 16 * 64];   // [l_time | l_fea]
  __shared__ _Float16 s_z3[4 * 16 * 128];  // [label | u_b]
  __shared__ float    s_red[4 * 16 * 16];
  const int tid = threadIdx.x, wave = tid >> 5, lane = tid & 31;
  const int half = lane >> 4, lm = lane & 15;
  _Float16* z2 = s_z2 + wave * 1024;
  _Float16* z3 = s_z3 + wave * 2048;
  float* red = s_red + wave * 256;

  stage_frags(s_f1, W_ll1, nullptr, 145, 0, 2, 5, tid, 128);
  stage_frags(s_f2, W_ll2, nullptr, 64, 0, 4, 2, tid, 128);
  stage_frags(s_fa, W_a1,  nullptr, 128, 0, 1, 4, tid, 128);

  const int R0 = (blockIdx.x * 4 + wave) * 16;

  {  // z2 cols [0,32) = l_time
    const float* lt = l_time + (R0 + lm) * 32 + 16 * half;
    _Float16* zr = z2 + lm * 64 + 16 * half;
#pragma unroll
    for (int i = 0; i < 16; ++i) zr[i] = (_Float16)lt[i];
  }
  {  // z3 cols [64,128) = broadcast unlabel[b], b = row/20
    int b = (R0 + lm) / 20;
    const float* up = unlab + (long)b * 64 + 32 * half;
    _Float16* zr = z3 + lm * 128 + 64 + 32 * half;
#pragma unroll
    for (int i = 0; i < 32; ++i) zr[i] = (_Float16)up[i];
  }
  __syncthreads();

  {  // l_fea = relu(feat @ W_ll1^T + b) -> z2 cols [32,64)
    const float* fr = feat + (long)(R0 + lm) * 145;
    v16h A[5];
#pragma unroll
    for (int kc = 0; kc < 5; ++kc) A[kc] = fragA_glb(fr, half, kc, 145);
#pragma unroll
    for (int ntb = 0; ntb < 2; ++ntb) {
      v8f c = gemm_tile<5>(s_f1, A, ntb, b_ll1[ntb * 16 + lm], lane);
      int j = lm + 16 * ntb;
#pragma unroll
      for (int r = 0; r < 8; ++r)
        z2[(r + 8 * half) * 64 + 32 + j] = (_Float16)fmaxf(c[r], 0.f);
    }
  }
  __syncthreads();

  {  // label = relu(z2 @ W_ll2^T + b) -> global + z3 cols [0,64)
    v16h A[2];
#pragma unroll
    for (int kc = 0; kc < 2; ++kc) A[kc] = fragA_lds(z2, 64, lm, half, kc);
#pragma unroll
    for (int ntb = 0; ntb < 4; ++ntb) {
      v8f c = gemm_tile<2>(s_f2, A, ntb, b_ll2[ntb * 16 + lm], lane);
      int j = lm + 16 * ntb;
#pragma unroll
      for (int r = 0; r < 8; ++r) {
        int m = r + 8 * half;
        float v = fmaxf(c[r], 0.f);
        label_out[(long)(R0 + m) * 64 + j] = v;
        z3[m * 128 + j] = (_Float16)v;
      }
    }
  }
  __syncthreads();

  {  // a1 = relu(z3 @ W_a1^T + b_a1); a = a1 @ W_a2^T + b_a2
    v16h A[4];
#pragma unroll
    for (int kc = 0; kc < 4; ++kc) A[kc] = fragA_lds(z3, 128, lm, half, kc);
    v8f c = gemm_tile<4>(s_fa, A, 0, b_a1[lm], lane);
#pragma unroll
    for (int r = 0; r < 8; ++r)
      red[(r + 8 * half) * 16 + lm] = fmaxf(c[r], 0.f);
  }
  __syncthreads();
  if (lane < 16) {
    float s = b_a2[0];
#pragma unroll
    for (int k = 0; k < 16; ++k) s += red[lane * 16 + k] * W_a2[k];
    a_out[R0 + lane] = s;
  }
}

// ============================================================================
// Kernel 4: IDW softmax over N=20 + weighted pooling sp (relu).
// ============================================================================
__global__ __launch_bounds__(64) void softmax_sp(
    const float* __restrict__ a, const float* __restrict__ dis,
    const float* __restrict__ label, float* __restrict__ sp) {
  __shared__ float sc[20];
  int b = blockIdx.x, j = threadIdx.x;
  if (j < 20) sc[j] = a[b * 20 + j] * dis[b * 20 + j];
  __syncthreads();
  float mx = -1e30f;
#pragma unroll
  for (int n = 0; n < 20; ++n) mx = fmaxf(mx, sc[n]);
  float sum = 0.f, acc = 0.f;
#pragma unroll
  for (int n = 0; n < 20; ++n) {
    float w = __expf(sc[n] - mx);
    sum += w;
    acc += w * label[((long)b * 20 + n) * 64 + j];
  }
  sp[(long)b * 64 + j] = fmaxf(acc * __builtin_amdgcn_rcpf(sum), 0.f);
}

// ============================================================================
// Kernel 5: GRU loop (hidden into cell is 0 each call => Whh_g reduces to
// biases) + final linear head.  X-stage = [unlabel(64) | temp(32) | sp(64)].
// ============================================================================
__global__ __launch_bounds__(128) void gru_out_kernel(
    const float* __restrict__ unlab, const float* __restrict__ sp,
    const float* __restrict__ h_in,
    const float* __restrict__ Wih_g, const float* __restrict__ bih_g,
    const float* __restrict__ bhh_g,
    const float* __restrict__ W_t, const float* __restrict__ b_t,
    const float* __restrict__ W_o, const float* __restrict__ b_o,
    float* __restrict__ out) {
  __shared__ _Float16 s_ft[2 * 512];       // W_t : K=32 (1kc) x 2nt
  __shared__ _Float16 s_fg[30 * 512];      // Wih_g: K=160 (5kc) x 6nt
  __shared__ _Float16 s_zx[4 * 16 * 160];  // X stage
  __shared__ _Float16 s_zh[4 * 16 * 32];   // h stage
  const int tid = threadIdx.x, wave = tid >> 5, lane = tid & 31;
  const int half = lane >> 4, lm = lane & 15;
  _Float16* zx = s_zx + wave * 2560;
  _Float16* zh = s_zh + wave * 512;

  stage_frags(s_ft, W_t,   nullptr, 32,  0, 2, 1, tid, 128);
  stage_frags(s_fg, Wih_g, nullptr, 160, 0, 6, 5, tid, 128);

  const int R0 = (blockIdx.x * 4 + wave) * 16;
  {
    const float* up = unlab + (long)(R0 + lm) * 64;
    const float* pp = sp    + (long)(R0 + lm) * 64;
    _Float16* zr = zx + lm * 160;
    for (int i = half; i < 64; i += 2) {
      zr[i]      = (_Float16)up[i];
      zr[96 + i] = (_Float16)pp[i];
    }
    const float* hp = h_in + (R0 + lm) * 32 + 16 * half;
    _Float16* hr = zh + lm * 32 + 16 * half;
#pragma unroll
    for (int i = 0; i < 16; ++i) hr[i] = (_Float16)hp[i];
  }
  float br[2], bz[2], bin[2], bhn[2];
#pragma unroll
  for (int ntb = 0; ntb < 2; ++ntb) {
    int j = lm + 16 * ntb;
    br[ntb]  = bih_g[j]      + bhh_g[j];
    bz[ntb]  = bih_g[32 + j] + bhh_g[32 + j];
    bin[ntb] = bih_g[64 + j];
    bhn[ntb] = bhh_g[64 + j];
  }
  __syncthreads();

  for (int it = 0; it < 6; ++it) {
    // temp = relu(h @ W_t^T + b_t) -> X cols [64,96)
    v16h Ah = fragA_lds(zh, 32, lm, half, 0);
#pragma unroll
    for (int ntb = 0; ntb < 2; ++ntb) {
      v8f c;
#pragma unroll
      for (int r = 0; r < 8; ++r) c[r] = b_t[ntb * 16 + lm];
      c = wmma16(Ah, fragB_lds(s_ft, ntb, lane), c);
      int j = lm + 16 * ntb;
#pragma unroll
      for (int r = 0; r < 8; ++r)
        zx[(r + 8 * half) * 160 + 64 + j] = (_Float16)fmaxf(c[r], 0.f);
    }
    __syncthreads();
    // gi = X @ Wih_g^T; r/z/n gate cols [0,32)/[32,64)/[64,96)
    v16h Ax[5];
#pragma unroll
    for (int kc = 0; kc < 5; ++kc) Ax[kc] = fragA_lds(zx, 160, lm, half, kc);
#pragma unroll
    for (int ntb = 0; ntb < 2; ++ntb) {
      v8f cr = gemm_tile<5>(s_fg, Ax, ntb,     br[ntb],  lane);
      v8f cz = gemm_tile<5>(s_fg, Ax, 2 + ntb, bz[ntb],  lane);
      v8f cn = gemm_tile<5>(s_fg, Ax, 4 + ntb, bin[ntb], lane);
      int j = lm + 16 * ntb;
#pragma unroll
      for (int r = 0; r < 8; ++r) {
        int m = r + 8 * half;
        float rv = sigf(cr[r]), zv = sigf(cz[r]);
        float nv = tanh_fast(cn[r] + rv * bhn[ntb]);
        zh[m * 32 + j] = (_Float16)((1.f - zv) * nv);  // h = (1-z)*n
      }
    }
    __syncthreads();
  }
  if (lane < 16) {
    float s = b_o[0];
#pragma unroll
    for (int k = 0; k < 32; ++k) s += (float)zh[lane * 32 + k] * W_o[k];
    out[R0 + lane] = s;
  }
}

// ============================================================================
extern "C" void kernel_launch(void* const* d_in, const int* in_sizes, int n_in,
                              void* d_out, int out_size, void* d_ws,
                              size_t ws_size, hipStream_t stream) {
  (void)in_sizes; (void)n_in; (void)out_size; (void)ws_size;
  const float* meo_u  = (const float*)d_in[0];
  const float* feat_u = (const float*)d_in[1];
  const float* ovi_l  = (const float*)d_in[2];
  const float* meo_l  = (const float*)d_in[3];
  const float* feat_l = (const float*)d_in[4];
  const float* dis_l  = (const float*)d_in[5];
  const float* h_t    = (const float*)d_in[6];
  const float* Wih_u  = (const float*)d_in[7];
  const float* Whh_u  = (const float*)d_in[8];
  const float* bih_u  = (const float*)d_in[9];
  const float* bhh_u  = (const float*)d_in[10];
  const float* Wih_l  = (const float*)d_in[11];
  const float* Whh_l  = (const float*)d_in[12];
  const float* bih_l  = (const float*)d_in[13];
  const float* bhh_l  = (const float*)d_in[14];
  const float* W_ul1  = (const float*)d_in[15];
  const float* b_ul1  = (const float*)d_in[16];
  const float* W_ul2  = (const float*)d_in[17];
  const float* b_ul2  = (const float*)d_in[18];
  const float* W_ll1  = (const float*)d_in[19];
  const float* b_ll1  = (const float*)d_in[20];
  const float* W_ll2  = (const float*)d_in[21];
  const float* b_ll2  = (const float*)d_in[22];
  const float* W_a1   = (const float*)d_in[23];
  const float* b_a1   = (const float*)d_in[24];
  const float* W_a2   = (const float*)d_in[25];
  const float* b_a2   = (const float*)d_in[26];
  const float* Wih_g  = (const float*)d_in[27];
  // d_in[28] (Whh_g) is mathematically unused: GRUCell hidden is zero.
  const float* bih_g  = (const float*)d_in[29];
  const float* bhh_g  = (const float*)d_in[30];
  const float* W_t    = (const float*)d_in[31];
  const float* b_t    = (const float*)d_in[32];
  const float* W_o    = (const float*)d_in[33];
  const float* b_o    = (const float*)d_in[34];
  // d_in[35] timestamp, d_in[36] label_id_list: unused by the forward math.

  constexpr int B = 8192, N = 20, T = 48;
  constexpr int RL = B * N;  // 163840 labeled rows

  // workspace layout (floats), ~69 MB total
  float* ws     = (float*)d_ws;
  float* u_time = ws;                          // B*32
  float* l_time = u_time + (size_t)B * 32;     // RL*32
  float* unlab  = l_time + (size_t)RL * 32;    // B*64
  float* label  = unlab + (size_t)B * 64;      // RL*64
  float* a_buf  = label + (size_t)RL * 64;     // RL
  float* sp_buf = a_buf + (size_t)RL;          // B*64

  // 1) both LSTMs (same kernel; labeled passes ovi, D=7)
  lstm_wmma<<<B / 128, 256, 0, stream>>>(nullptr, meo_u, Wih_u, Whh_u, bih_u,
                                         bhh_u, u_time, 6, T);
  lstm_wmma<<<RL / 128, 256, 0, stream>>>(ovi_l, meo_l, Wih_l, Whh_l, bih_l,
                                          bhh_l, l_time, 7, T);
  // 2) unlabeled feature branch
  unlabeled_branch<<<B, 64, 0, stream>>>(feat_u, u_time, W_ul1, b_ul1, W_ul2,
                                         b_ul2, unlab);
  // 3) labeled fusion + attention logits
  labeled_fusion<<<RL / 64, 128, 0, stream>>>(feat_l, l_time, unlab, W_ll1,
                                              b_ll1, W_ll2, b_ll2, W_a1, b_a1,
                                              W_a2, b_a2, label, a_buf);
  // 4) softmax + pooled sp
  softmax_sp<<<B, 64, 0, stream>>>(a_buf, dis_l, label, sp_buf);
  // 5) GRU loop + output head
  gru_out_kernel<<<B / 64, 128, 0, stream>>>(unlab, sp_buf, h_t, Wih_g, bih_g,
                                             bhh_g, W_t, b_t, W_o, b_o,
                                             (float*)d_out);
}
